// Low_Fidelity_FusionDeepONet_16054587752612
// MI455X (gfx1250) — matrix operand order repaired and checked by reading, not compiled, not run on hardware
//
#include <hip/hip_runtime.h>
#include <hip/hip_bf16.h>

// ---------------------------------------------------------------------------
// FusionDeepONet trunk on MI455X (gfx1250): f16 WMMA 16x16x32 with f32 accum.
// B=16, NPTS=20000, H=128, O=4, L=6, PDIM=8, CDIM=3, K=3 (rowdy sine terms)
// ---------------------------------------------------------------------------

typedef __attribute__((ext_vector_type(16))) _Float16 v16h;
typedef __attribute__((ext_vector_type(8)))  _Float16 v8h;
typedef __attribute__((ext_vector_type(8)))  float    v8f;

#define HDIM   128
#define LNUM   6
#define NPTSC  20000
#define BATCH  16
#define ODIM   4
#define ROWS   64      // points per block
#define PITCH  136     // halves per LDS row (128 + 8 pad, keeps 16B alignment)

// Branch-free tanh: single CDNA5 v_tanh_f32 TRANS op when available.
__device__ __forceinline__ float fast_tanh(float z) {
#if __has_builtin(__builtin_amdgcn_tanhf)
  return __builtin_amdgcn_tanhf(z);
#else
  float az = fabsf(z);
  float e  = __expf(-2.f * az);                      // v_exp_f32 path
  float t  = (1.f - e) * __builtin_amdgcn_rcpf(1.f + e);
  return copysignf(t, z);
#endif
}

// ---------------------------------------------------------------------------
// Kernel 1: convert trunk weights fp32 -> f16, column-major (Wt[n][k]).
// Layer 0 (K=4) padded to K=32 with zeros so it shares the WMMA path.
// ---------------------------------------------------------------------------
__global__ void convert_kernel(const float* __restrict__ tW0,
                               const float* __restrict__ tWr,
                               _Float16* __restrict__ Wt0,
                               _Float16* __restrict__ Wtr) {
  int tid = blockIdx.x * blockDim.x + threadIdx.x;
  if (tid < HDIM * 32) {
    int n = tid >> 5, k = tid & 31;
    Wt0[tid] = (k < 4) ? (_Float16)tW0[k * HDIM + n] : (_Float16)0.f;
  } else {
    int t = tid - HDIM * 32;
    if (t < (LNUM - 1) * HDIM * HDIM) {
      int i = t / (HDIM * HDIM), r = t % (HDIM * HDIM);
      int n = r >> 7, k = r & 127;
      Wtr[t] = (_Float16)tWr[(i * HDIM + k) * HDIM + n];
    }
  }
}

// ---------------------------------------------------------------------------
// Kernel 2: branch tower (tiny: B=16). Produces cumulative skips S[L][B][H],
// and the FUSED final projection Wcomb[b] = final_W @ ZL[b]^T (128x4 per
// batch) + bias_out[b][o] = final_b . ZL[b][o]. This deletes the per-point
// 128x128 final GEMM entirely.
// ---------------------------------------------------------------------------
__global__ void branch_kernel(const float* __restrict__ params,
                              const float* __restrict__ bW0,
                              const float* __restrict__ bWr,
                              const float* __restrict__ bb,
                              const float* __restrict__ bWout,
                              const float* __restrict__ bbout,
                              const float* __restrict__ fW,
                              const float* __restrict__ fb,
                              float* __restrict__ Sout,
                              float* __restrict__ Wcomb,
                              float* __restrict__ biasOut) {
  __shared__ float hbuf[BATCH][HDIM];
  __shared__ float zl[BATCH][HDIM * ODIM];
  const int j = threadIdx.x;  // 0..127 : one H channel per thread

  // layer 0
  for (int b = 0; b < BATCH; ++b) {
    float acc = bb[j];
#pragma unroll
    for (int p = 0; p < 8; ++p) acc += params[b * 8 + p] * bW0[p * HDIM + j];
    float hv = fast_tanh(acc);
    hbuf[b][j] = hv;
    Sout[b * HDIM + j] = hv;
  }
  __syncthreads();

  for (int i = 1; i < LNUM; ++i) {
    float hn[BATCH];
    for (int b = 0; b < BATCH; ++b) {
      float acc = bb[i * HDIM + j];
      for (int k = 0; k < HDIM; ++k)
        acc += hbuf[b][k] * bWr[((i - 1) * HDIM + k) * HDIM + j];
      hn[b] = fast_tanh(acc);
    }
    __syncthreads();
    for (int b = 0; b < BATCH; ++b) {
      hbuf[b][j] = hn[b];
      Sout[(i * BATCH + b) * HDIM + j] =
          Sout[((i - 1) * BATCH + b) * HDIM + j] + hn[b];
    }
    __syncthreads();
  }

  // branch_out -> ZL  (B x O x H laid out as zl[b][o*H + h])
  for (int b = 0; b < BATCH; ++b)
    for (int c = j; c < HDIM * ODIM; c += HDIM) {
      float acc = bbout[c];
      for (int k = 0; k < HDIM; ++k) acc += hbuf[b][k] * bWout[k * (HDIM * ODIM) + c];
      zl[b][c] = acc;
    }
  __syncthreads();

  // Wcomb[b][k=j][o] = sum_h final_W[j][h] * ZL[b][o][h]
  for (int b = 0; b < BATCH; ++b)
    for (int o = 0; o < ODIM; ++o) {
      float acc = 0.f;
      for (int k = 0; k < HDIM; ++k) acc += fW[j * HDIM + k] * zl[b][o * HDIM + k];
      Wcomb[(b * HDIM + j) * ODIM + o] = acc;
    }
  if (j < BATCH * ODIM) {
    int b = j >> 2, o = j & 3;
    float acc = 0.f;
    for (int k = 0; k < HDIM; ++k) acc += fb[k] * zl[b][o * HDIM + k];
    biasOut[b * ODIM + o] = acc;
  }
}

// ---------------------------------------------------------------------------
// Kernel 3: trunk tower. 256 threads = 8 wave32; wave w owns N-tile w (16 of
// 128 channels); block owns 64 points (4 M-tiles). Layers 1-5: 4 unrolled
// K-steps x 4 M-tiles = 16 v_wmma_f32_16x16x32_f16 per wave per layer, with
// the B fragment reused across M-tiles. Rowdy activation = v_tanh + v_sin +
// v_cos (TRANS32, co-executes with WMMA), f16 ping-pong through LDS.
// ---------------------------------------------------------------------------
__global__ __launch_bounds__(256)
void trunk_kernel(const float* __restrict__ coords, const float* __restrict__ sdf,
                  const float* __restrict__ trunk_b, const float* __restrict__ rowdy_a,
                  const _Float16* __restrict__ Wt0, const _Float16* __restrict__ Wtr,
                  const float* __restrict__ Sbuf, const float* __restrict__ Wcomb,
                  const float* __restrict__ biasOut, float* __restrict__ out) {
  __shared__ __align__(32) _Float16 xbuf[2][ROWS * PITCH];  // ~34 KB
  __shared__ float wcs[HDIM * ODIM];

  const int tid = threadIdx.x;
  const int lane = tid & 31;
  const int wave = tid >> 5;       // N-tile index 0..7
  const int b = blockIdx.y;
  const int p0 = blockIdx.x * ROWS;
  const int hiHalf = lane >> 4;    // 0: lanes 0-15, 1: lanes 16-31

  // ---- phase 0: load [coords|sdf] padded to K=32 into xbuf[0] -------------
  for (int idx = tid; idx < ROWS * 32; idx += 256) {
    int row = idx >> 5, col = idx & 31;
    int pt = p0 + row; if (pt >= NPTSC) pt = NPTSC - 1;
    float v = 0.f;
    if (col < 3)       v = coords[((long)b * NPTSC + pt) * 3 + col];
    else if (col == 3) v = sdf[(long)b * NPTSC + pt];
    xbuf[0][row * PITCH + col] = (_Float16)v;
  }
  __syncthreads();

  const int h = wave * 16 + (lane & 15);  // this lane's H channel (col)
  int cur = 0;

  // one 32-wide K step: B frag (global, L2-hot) x 4 A frags (LDS)
  auto kstep = [&](const _Float16* __restrict__ W, int kw, int s, v8f* acc) {
    const _Float16* bp = W + h * kw + s * 32 + (hiHalf << 4);
    v16h bfrag = *reinterpret_cast<const v16h*>(bp);
    const _Float16* abase =
        &xbuf[cur][(lane & 15) * PITCH + s * 32 + (hiHalf << 4)];
#pragma unroll
    for (int m = 0; m < 4; ++m) {
      const _Float16* ap = abase + m * 16 * PITCH;
      v8h lo = *reinterpret_cast<const v8h*>(ap);
      v8h hi = *reinterpret_cast<const v8h*>(ap + 8);
      v16h afrag = __builtin_shufflevector(lo, hi, 0, 1, 2, 3, 4, 5, 6, 7,
                                           8, 9, 10, 11, 12, 13, 14, 15);
      acc[m] = __builtin_amdgcn_wmma_f32_16x16x32_f16(
          false, afrag, false, bfrag, (short)0, acc[m], false, false);
    }
  };

  // rowdy activation + skip-scale on the f32 accumulators, store f16
  auto activate = [&](int i, v8f* acc) {
    const float bias = trunk_b[i * HDIM + h];
    const float a1 = rowdy_a[(i * 3 + 0) * HDIM + h];
    const float a2 = rowdy_a[(i * 3 + 1) * HDIM + h];
    const float a3 = rowdy_a[(i * 3 + 2) * HDIM + h];
    const float Sv = Sbuf[(i * BATCH + b) * HDIM + h];
    _Float16* xn = xbuf[cur ^ 1];
#pragma unroll
    for (int m = 0; m < 4; ++m) {
#pragma unroll
      for (int r = 0; r < 8; ++r) {
        float z = acc[m][r] + bias;
        float sz = __sinf(z);
        float cz = __cosf(z);
        float s2 = 2.f * sz * cz;               // sin(2z)
        float s3 = sz * (3.f - 4.f * sz * sz);  // sin(3z)
        float v = fast_tanh(z) + a1 * sz + a2 * s2 + a3 * s3;
        v *= Sv;
        int row = m * 16 + r + (hiHalf << 3);   // C layout: VGPR r -> M=r/r+8
        xn[row * PITCH + h] = (_Float16)v;
      }
    }
  };

  // ---- layer 0: K padded to 32, single K-step --------------------------------
  {
    v8f acc[4];
#pragma unroll
    for (int m = 0; m < 4; ++m)
#pragma unroll
      for (int r = 0; r < 8; ++r) acc[m][r] = 0.f;
    kstep(Wt0, 32, 0, acc);
    activate(0, acc);
    __syncthreads();
    cur ^= 1;
  }

  // ---- layers 1..5: K = 128, 4 unrolled K-steps ------------------------------
  for (int i = 1; i < LNUM; ++i) {
    const _Float16* W = Wtr + (i - 1) * HDIM * HDIM;
    v8f acc[4];
#pragma unroll
    for (int m = 0; m < 4; ++m)
#pragma unroll
      for (int r = 0; r < 8; ++r) acc[m][r] = 0.f;
#pragma unroll
    for (int s = 0; s < 4; ++s) kstep(W, HDIM, s, acc);
    activate(i, acc);
    __syncthreads();
    cur ^= 1;
  }

  // ---- fused final: out[b][p][o] = x[p] . Wcomb[b][:,o] + bias_out[b][o] ----
  for (int idx = tid; idx < HDIM * ODIM; idx += 256)
    wcs[idx] = Wcomb[b * HDIM * ODIM + idx];
  __syncthreads();

  const int p = tid >> 2;
  const int o = tid & 3;
  const _Float16* xr = &xbuf[cur][p * PITCH];
  float acco = biasOut[b * ODIM + o];
#pragma unroll 8
  for (int k = 0; k < HDIM; ++k) acco += (float)xr[k] * wcs[k * ODIM + o];
  int pt = p0 + p;
  if (pt < NPTSC) out[((long)b * NPTSC + pt) * ODIM + o] = acco;
}

// ---------------------------------------------------------------------------
extern "C" void kernel_launch(void* const* d_in, const int* in_sizes, int n_in,
                              void* d_out, int out_size, void* d_ws, size_t ws_size,
                              hipStream_t stream) {
  const float* coords = (const float*)d_in[0];
  const float* sdf    = (const float*)d_in[1];
  const float* params = (const float*)d_in[2];
  const float* bW0    = (const float*)d_in[3];
  const float* bWr    = (const float*)d_in[4];
  const float* bb     = (const float*)d_in[5];
  const float* bWout  = (const float*)d_in[6];
  const float* bbout  = (const float*)d_in[7];
  const float* tW0    = (const float*)d_in[8];
  const float* tWr    = (const float*)d_in[9];
  const float* tb     = (const float*)d_in[10];
  const float* ra     = (const float*)d_in[11];
  const float* fW     = (const float*)d_in[12];
  const float* fb     = (const float*)d_in[13];
  float* out = (float*)d_out;

  // workspace layout (bytes)
  char* ws = (char*)d_ws;
  _Float16* Wt0 = (_Float16*)(ws + 0);        // 128*32 halves    (8192 B)
  _Float16* Wtr = (_Float16*)(ws + 8192);     // 5*128*128 halves (163840 B)
  float* Sbuf   = (float*)(ws + 172032);      // 6*16*128 f32     (49152 B)
  float* WcombG = (float*)(ws + 221184);      // 16*128*4 f32     (32768 B)
  float* biasO  = (float*)(ws + 253952);      // 16*4 f32         (256 B)

  const int convN = HDIM * 32 + (LNUM - 1) * HDIM * HDIM;
  convert_kernel<<<(convN + 255) / 256, 256, 0, stream>>>(tW0, tWr, Wt0, Wtr);
  branch_kernel<<<1, HDIM, 0, stream>>>(params, bW0, bWr, bb, bWout, bbout,
                                        fW, fb, Sbuf, WcombG, biasO);
  dim3 grid((NPTSC + ROWS - 1) / ROWS, BATCH);
  trunk_kernel<<<grid, 256, 0, stream>>>(coords, sdf, tb, ra, Wt0, Wtr,
                                         Sbuf, WcombG, biasO, out);
}
